// ToxicityLSTM_6451040878612
// MI455X (gfx1250) — matrix operand order repaired and checked by reading, not compiled, not run on hardware
//
#include <hip/hip_runtime.h>
#include <hip/hip_bf16.h>

#define T_STEPS 1024
#define BATCH   64
#define HID     256
#define KDIM    512          // D + H
#define BH      (BATCH * HID)
#define AROW    264          // LDS row stride in halves: 256 data + 8 pad (bank-conflict free)

typedef __attribute__((ext_vector_type(16))) __bf16 v16bf;
typedef __attribute__((ext_vector_type(8)))  float  v8f;
typedef __attribute__((ext_vector_type(4)))  unsigned int u32x4;
typedef __attribute__((ext_vector_type(8)))  int          i32x8;
typedef __attribute__((ext_vector_type(4)))  int          i32x4;

// global-address-space pointer: keeps laundered weight loads on the
// global_load path (own LOADcnt) instead of degrading to flat_load.
typedef const __attribute__((address_space(1))) u32x4* gq4;

#if defined(__gfx1250__) && __has_builtin(__builtin_amdgcn_tensor_load_to_lds) && \
    __has_builtin(__builtin_amdgcn_s_wait_tensorcnt)
#define HAVE_TDM 1
#else
#define HAVE_TDM 0
#endif

#if __has_builtin(__builtin_amdgcn_sched_group_barrier)
#define SGB(mask, n, sync) __builtin_amdgcn_sched_group_barrier(mask, n, sync)
#else
#define SGB(mask, n, sync)
#endif

union Frag16 {
    v16bf v;
    unsigned short u[16];
    u32x4 q[2];
};

__device__ __forceinline__ unsigned short f2bf(float f) {
    unsigned int u = __builtin_bit_cast(unsigned int, f);
    unsigned int r = u + 0x7FFFu + ((u >> 16) & 1u);   // round-to-nearest-even
    return (unsigned short)(r >> 16);
}
__device__ __forceinline__ float bf2f(unsigned short h) {
    unsigned int u = ((unsigned int)h) << 16;
    return __builtin_bit_cast(float, u);
}
__device__ __forceinline__ float sigm(float x) { return 1.f / (1.f + __expf(-x)); }
__device__ __forceinline__ float tanh_(float x) {
    float c = fminf(fmaxf(x, -15.f), 15.f);
    float e = __expf(2.f * c);
    return (e - 1.f) / (e + 1.f);
}
__device__ __forceinline__ v8f wmma_bf16(v16bf a, v16bf b, v8f c) {
    return __builtin_amdgcn_wmma_f32_16x16x32_bf16(
        false, a, false, b, (short)0, c, false, false);
}

#if HAVE_TDM
// One TDM instruction: 32 KB (16384 x 2B) global -> LDS, inserting 16 bytes of
// LDS padding after every 512-byte row so rows land 4 banks apart.
// D# group0: count=1, lds_addr, global_addr, type=2 ("image").
// D# group1: data_size=2B; tensor_dim0 = tile_dim0 = stride = 16384; dim1=1;
//            pad_enable=1, pad_interval=6 (every 128 DWORDs), pad_amount=3 (4 DWORDs).
__device__ __forceinline__ void tdm_load_rows(const unsigned short* gsrc, unsigned short* ldst) {
    unsigned long long ga = (unsigned long long)(uintptr_t)gsrc;
    u32x4 g0;
    g0[0] = 1u;                                     // count = 1 valid descriptor
    g0[1] = (unsigned int)(uintptr_t)ldst;          // LDS byte address
    g0[2] = (unsigned int)ga;                       // global_addr[31:0]
    g0[3] = ((unsigned int)(ga >> 32) & 0x01FFFFFFu) | 0x80000000u;  // addr[56:32] | type=2
    i32x8 g1 = { 0x07910000,                        // data_size=2B | pad_en | int=6 | amt=3
                 (int)0x40000000,                   // tensor_dim0[15:0]=16384 in [31:16]
                 0x00010000,                        // tensor_dim1 = 1
                 (int)0x40000000,                   // tile_dim0 = 16384 in [31:16]
                 0,                                 // tile_dim1/2 = 0 (unused)
                 16384,                             // tensor_dim0_stride = 16384
                 0, 0 };
    i32x4 z4 = {0, 0, 0, 0};
    i32x8 z8 = {0, 0, 0, 0, 0, 0, 0, 0};
    __builtin_amdgcn_tensor_load_to_lds(g0, g1, z4, z4, z8, 0);
}
#endif

// 4-block synchronization: CDNA5 cluster split-barrier (NOP when dispatch is
// not cluster-configured) + monotonic L2 atomic barrier as the safe fallback.
__device__ __forceinline__ void blocks4_sync(int* bar, int target) {
    __threadfence();
    __syncthreads();
    if (threadIdx.x == 0) {
        __hip_atomic_fetch_add(bar, 1, __ATOMIC_ACQ_REL, __HIP_MEMORY_SCOPE_AGENT);
        while (__hip_atomic_load(bar, __ATOMIC_ACQUIRE, __HIP_MEMORY_SCOPE_AGENT) < target) {
            __builtin_amdgcn_s_sleep(1);
        }
    }
    if (threadIdx.x < 32) {                       // one wave signals per workgroup
        asm volatile("s_barrier_signal -3" ::: "memory");
    }
    asm volatile("s_barrier_wait -3" ::: "memory");
    __syncthreads();
}

// Swizzle f32 kernel [512][1024] -> fragment-major bf16:
// wfrag[((colblk*16 + kblk)*32 + lane)*16 + j], col = colblk*16+lane%16,
// k = kblk*32 + (j/8)*16 + (lane/16)*8 + j%8  -> each wave's B-fragment is a 1KB burst.
__global__ void convert_weights_kernel(const float* __restrict__ kern,
                                       unsigned short* __restrict__ wfrag) {
    int o = blockIdx.x * blockDim.x + threadIdx.x;
    if (o >= KDIM * 1024) return;
    int j      = o & 15;
    int lane   = (o >> 4) & 31;
    int kblk   = (o >> 9) & 15;
    int colblk = o >> 13;
    int col = colblk * 16 + (lane & 15);
    int k   = kblk * 32 + ((j >> 3) << 4) + ((lane >> 4) << 3) + (j & 7);
    wfrag[o] = f2bf(kern[k * 1024 + col]);
}

__global__ void init_state_kernel(unsigned short* h0slot0, unsigned short* h1slot0, int* bar) {
    int i = blockIdx.x * blockDim.x + threadIdx.x;
    if (i < BH) { h0slot0[i] = 0; h1slot0[i] = 0; }
    if (i < 2)  bar[i] = 0;
}

// Persistent LSTM layer: 4 blocks (one WGP each) x 512 threads (16 waves).
// Block owns 64 hidden columns; wave owns one 16x16 (batch x hidden) tile and
// its persistent cell state; computes 4 gate accumulators over K=512.
template <int LAYER>
__global__ void __launch_bounds__(512)
lstm_layer_kernel(const float* __restrict__ x,
                  const unsigned short* __restrict__ xseq,
                  const unsigned short* __restrict__ wfrag,
                  const float* __restrict__ bias,
                  unsigned short* __restrict__ hseq,
                  unsigned short* __restrict__ hping,
                  int* __restrict__ bar) {
    __shared__ unsigned short a_x[BATCH * AROW];   // x_t     bf16, padded rows
    __shared__ unsigned short a_h[BATCH * AROW];   // h_{t-1} bf16, padded rows

    const int tid   = threadIdx.x;
    const int lane  = tid & 31;
    const int wave  = tid >> 5;        // 16 waves
    const int chunk = blockIdx.x;      // hidden cols [chunk*64, chunk*64+64)
    const int mt    = wave >> 2;       // batch tile 0..3
    const int nt    = wave & 3;        // hidden tile 0..3

    const v8f vzero = {0.f, 0.f, 0.f, 0.f, 0.f, 0.f, 0.f, 0.f};
    v8f cst = vzero;                   // persistent cell state (16x16 tile)

    // loop-invariant fragment base pointers (immediate offsets per kblk)
    const int arow = (mt * 16 + (lane & 15)) * AROW + ((lane >> 4) << 3);
    const u32x4* abx = (const u32x4*)(a_x + arow);
    const u32x4* abh = (const u32x4*)(a_h + arow);
    const u32x4* bb[4];
#pragma unroll
    for (int g = 0; g < 4; ++g) {
        int colblk = (g << 4) + (chunk << 2) + nt;
        bb[g] = (const u32x4*)(wfrag + ((size_t)colblk * 512 + lane) * 16);
    }
    float bgate[4];
    {
        int col = chunk * 64 + nt * 16 + (lane & 15);
#pragma unroll
        for (int g = 0; g < 4; ++g) bgate[g] = bias[g * HID + col];
    }

#pragma unroll 1
    for (int t = 0; t < T_STEPS; ++t) {
        // ---- stage A = [x_t | h_{t-1}] into LDS (bf16, padded rows) ----
        if (LAYER == 0) {
            const unsigned short* hp = hseq + (size_t)t * BH;
#if HAVE_TDM
            if (wave == 0) {                       // DMA 32KB h tile, EXEC-independent
                tdm_load_rows(hp, a_h);
                __builtin_amdgcn_s_wait_tensorcnt(0);
            }
#else
            for (int i = tid; i < BH; i += 512) a_h[(i >> 8) * AROW + (i & 255)] = hp[i];
#endif
            for (int i = tid; i < BH; i += 512) {  // f32 -> bf16 convert of x_t
                int b = i >> 8, d = i & 255;
                size_t xoff = (size_t)b * T_STEPS * HID + (size_t)t * HID + d;
                a_x[b * AROW + d] = f2bf(x[xoff]);
                if (t + 1 < T_STEPS) __builtin_prefetch(&x[xoff + HID], 0, 0);
            }
        } else {
            const unsigned short* xs = xseq + (size_t)(t + 1) * BH;  // layer-0 h_t
            const unsigned short* hp = hping + (size_t)(t & 1) * BH;
#if HAVE_TDM
            if (wave == 0) {
                tdm_load_rows(xs, a_x);
                tdm_load_rows(hp, a_h);
                __builtin_amdgcn_s_wait_tensorcnt(0);
            }
#else
            for (int i = tid; i < BH; i += 512) {
                int b = i >> 8, d = i & 255;
                a_x[b * AROW + d] = xs[i];
                a_h[b * AROW + d] = hp[i];
            }
#endif
        }
        __syncthreads();

        // Launder B base pointers (as explicit global-AS pointers) so LLVM can
        // neither hoist the t-invariant weight loads out of the 1024-step loop
        // (register spills) nor demote them to flat_load.
        gq4 wb[4];
#pragma unroll
        for (int g = 0; g < 4; ++g) {
            wb[g] = (gq4)(uintptr_t)bb[g];
            asm volatile("" : "+v"(wb[g]));
        }

        // ---- gates = A @ W_slice : 4 gate accumulators, K = 512 ----
        // Software-pipelined with explicit scheduler directives: fragments for
        // kblk+1 are in flight while the 4 WMMAs of kblk execute.
        v8f acc[4] = {vzero, vzero, vzero, vzero};
        Frag16 af[2];
        Frag16 bf[2][4];
        af[0].q[0] = abx[0];
        af[0].q[1] = abx[2];
#pragma unroll
        for (int g = 0; g < 4; ++g) {
            bf[0][g].q[0] = wb[g][0];
            bf[0][g].q[1] = wb[g][1];
        }
#pragma unroll
        for (int kblk = 0; kblk < 16; ++kblk) {
            const int cur = kblk & 1, nxt = cur ^ 1;
            if (kblk < 15) {
                const int kn = kblk + 1;
                const u32x4* ab = (kn < 8) ? abx : abh;   // x half / h half of K
                const int kk = kn & 7;
                af[nxt].q[0] = ab[kk * 4];
                af[nxt].q[1] = ab[kk * 4 + 2];
#pragma unroll
                for (int g = 0; g < 4; ++g) {
                    bf[nxt][g].q[0] = wb[g][kn * 64];
                    bf[nxt][g].q[1] = wb[g][kn * 64 + 1];
                }
            }
#pragma unroll
            for (int g = 0; g < 4; ++g)
                acc[g] = wmma_bf16(af[cur].v, bf[cur][g].v, acc[g]);
        }
        // Pipeline schedule: prime stage-0 loads, then { stage k+1 loads ; 4 WMMA(k) }.
        // masks: 0x100 = DS read, 0x020 = VMEM read, 0x008 = MFMA/WMMA.
        SGB(0x100, 2, 0);
        SGB(0x020, 8, 0);
#pragma unroll
        for (int i = 0; i < 15; ++i) {
            SGB(0x100, 2, 0);
            SGB(0x020, 8, 0);
            SGB(0x008, 4, 0);
        }
        SGB(0x008, 4, 0);

        // ---- elementwise LSTM cell, publish h tile ----
        unsigned short* hout = (LAYER == 0)
            ? (hseq + (size_t)(t + 1) * BH)
            : (hping + (size_t)((t + 1) & 1) * BH);
        const int col = chunk * 64 + nt * 16 + (lane & 15);
#pragma unroll
        for (int v = 0; v < 8; ++v) {
            float gi = acc[0][v] + bgate[0];
            float gj = acc[1][v] + bgate[1];
            float gf = acc[2][v] + bgate[2];
            float go = acc[3][v] + bgate[3];
            float cv = sigm(gf + 1.0f) * cst[v] + sigm(gi) * tanh_(gj);
            float hv = sigm(go) * tanh_(cv);
            cst[v] = cv;
            int row = mt * 16 + ((lane < 16) ? v : v + 8);   // C/D VGPR layout
            hout[row * HID + col] = f2bf(hv);
        }

        blocks4_sync(bar, 4 * (t + 1));
    }
}

// last_h [64,256] bf16 @ w_out [256,6] + b_out -> logits / sigmoid / threshold
__global__ void proj_kernel(const unsigned short* __restrict__ h,
                            const float* __restrict__ w_out,
                            const float* __restrict__ b_out,
                            float* __restrict__ out) {
    int tid = threadIdx.x;
    if (tid >= BATCH * 6) return;
    int b = tid / 6, c = tid % 6;
    float s = b_out[c];
    for (int k = 0; k < HID; ++k) s += bf2f(h[b * HID + k]) * w_out[k * 6 + c];
    float sg = 1.f / (1.f + __expf(-s));
    out[tid]       = s;                        // logits
    out[384 + tid] = sg;                       // output
    out[768 + tid] = (sg > 0.5f) ? 1.f : 0.f;  // prediction
}

extern "C" void kernel_launch(void* const* d_in, const int* in_sizes, int n_in,
                              void* d_out, int out_size, void* d_ws, size_t ws_size,
                              hipStream_t stream) {
    const float* x       = (const float*)d_in[0];
    const float* kernel0 = (const float*)d_in[1];
    const float* bias0   = (const float*)d_in[2];
    const float* kernel1 = (const float*)d_in[3];
    const float* bias1   = (const float*)d_in[4];
    const float* w_out   = (const float*)d_in[5];
    const float* b_out   = (const float*)d_in[6];
    (void)in_sizes; (void)n_in; (void)out_size; (void)ws_size;

    char* ws = (char*)d_ws;
    size_t off = 0;
    auto take = [&](size_t bytes) -> char* {
        char* p = ws + off;
        off += (bytes + 255) & ~(size_t)255;
        return p;
    };
    unsigned short* wf0   = (unsigned short*)take((size_t)KDIM * 1024 * 2);        // 1 MB
    unsigned short* wf1   = (unsigned short*)take((size_t)KDIM * 1024 * 2);        // 1 MB
    unsigned short* h0seq = (unsigned short*)take((size_t)(T_STEPS + 1) * BH * 2); // ~33.6 MB
    unsigned short* h1p   = (unsigned short*)take((size_t)2 * BH * 2);             // 64 KB
    int*            bar   = (int*)take(256);

    convert_weights_kernel<<<2048, 256, 0, stream>>>(kernel0, wf0);
    convert_weights_kernel<<<2048, 256, 0, stream>>>(kernel1, wf1);
    init_state_kernel<<<64, 256, 0, stream>>>(h0seq, h1p, bar);

    lstm_layer_kernel<0><<<4, 512, 0, stream>>>(x, nullptr, wf0, bias0, h0seq, nullptr, bar);
    lstm_layer_kernel<1><<<4, 512, 0, stream>>>(nullptr, h0seq, wf1, bias1, nullptr, h1p, bar + 1);

    // after t = 1023 the final h of layer 1 sits in ping-pong slot 0
    proj_kernel<<<1, 384, 0, stream>>>(h1p, w_out, b_out, (float*)d_out);
}
